// BiMambaBlock_83674552861035
// MI455X (gfx1250) — matrix operand order repaired and checked by reading, not compiled, not run on hardware
//
#include <hip/hip_runtime.h>
#include <hip/hip_bf16.h>

// ---------------------------------------------------------------------------
// BiMamba block for MI455X (gfx1250, wave32, WMMA).
// GEMMs run on the matrix pipe via v_wmma_f32_16x16x32_bf16 with f32->bf16
// conversion during vectorized LDS staging. Sequence reversal for the backward
// direction is fused into the GEMM row remap (revA on input, revC on output).
// M is always a multiple of 128 and K a multiple of 32 in this pipeline, so
// only the N dimension carries bounds checks (for the N=96 projection).
// ---------------------------------------------------------------------------

typedef __attribute__((ext_vector_type(16))) __bf16    v16bf;
typedef __attribute__((ext_vector_type(4)))  __bf16    v4bf;
typedef __attribute__((ext_vector_type(8)))  float     v8f;
typedef __attribute__((ext_vector_type(4)))  float     v4f;
typedef __attribute__((ext_vector_type(4)))  unsigned  u32x4;

union ABfrag { v16bf v; u32x4 q[2]; };

#define BSZ     2
#define SEQL    1024
#define DMODEL  1024
#define DINNER  2048
#define DSTATE  16
#define DCONV   4
#define DTRANK  64
#define BL      (BSZ * SEQL)

#define TM 128
#define TN 128
#define TK 32

__device__ __forceinline__ float silu_f(float x) {
    return x / (1.0f + __expf(-x));
}

__device__ __forceinline__ int rev_row(int row) {
    // reverse l within each batch of SEQL rows (SEQL = 1024, power of two)
    int b = row >> 10;
    int l = row & (SEQL - 1);
    return (b << 10) + (SEQL - 1 - l);
}

// ---------------------------------------------------------------------------
// bf16-WMMA GEMM:  C[M,N] = epilogue(A[M,K] * B[K,N])
//   mode 0: plain   mode 1: + bias[col]   mode 2: softplus(acc + bias[col])
// Requires M % 128 == 0, K % 32 == 0. N may be ragged (zero-fill / skip).
// ---------------------------------------------------------------------------
__global__ __launch_bounds__(256)
void gemm_bf16_wmma(const float* __restrict__ A, const float* __restrict__ Bw,
                    float* __restrict__ C, const float* __restrict__ bias,
                    int M, int N, int K, int lda, int ldb, int ldc,
                    int col_off, int mode, int revA, int revC)
{
    __shared__ __align__(16) __bf16 sA[TM * TK];   // [m][k] row-major
    __shared__ __align__(16) __bf16 sB[TN * TK];   // [n][k] (transposed)

    const int tid  = threadIdx.x;
    const int lane = tid & 31;
    const int wid  = tid >> 5;
    const int block_m = blockIdx.y * TM;
    const int block_n = blockIdx.x * TN;
    const int wm = (wid & 3) * 32;   // wave's M offset inside tile (4 waves)
    const int wn = (wid >> 2) * 64;  // wave's N offset inside tile (2 waves)
    const int half = lane >> 4;      // 0: lanes 0-15, 1: lanes 16-31
    const int mn   = lane & 15;

    // Per-thread staging coordinates (hoisted out of the K loop).
    // A: 1024 float4 chunks; thread covers 4: row = lin>>3, kq = lin&7.
    int a_row[4], a_kq[4];
    #pragma unroll
    for (int e = 0; e < 4; ++e) {
        int lin = e * 256 + tid;
        int r   = lin >> 3;
        a_kq[e] = lin & 7;
        int grow = block_m + r;
        a_row[e] = revA ? rev_row(grow) : grow;
        (void)0;
    }
    // B: 1024 (n, kq) chunks; thread covers 4: n = lin&127, kq = lin>>7.
    int b_n[4], b_kq[4];
    #pragma unroll
    for (int e = 0; e < 4; ++e) {
        int lin = e * 256 + tid;
        b_n[e]  = lin & 127;
        b_kq[e] = lin >> 7;
    }

    v8f acc[2][4] = {};

    for (int k0 = 0; k0 < K; k0 += TK) {
        // Speculative prefetch of the next K-chunk (global_prefetch_b8).
        if (k0 + TK < K) {
            __builtin_prefetch(&A[(long)a_row[0] * lda + k0 + TK + a_kq[0] * 4], 0, 1);
            __builtin_prefetch(&Bw[(long)(k0 + TK + b_kq[0] * 4) * ldb + block_n + b_n[0]], 0, 1);
        }

        // ---- stage A tile: 128x32, vector loads + packed bf16 stores ----
        #pragma unroll
        for (int e = 0; e < 4; ++e) {
            const v4f f = *(const v4f*)&A[(long)a_row[e] * lda + k0 + a_kq[e] * 4];
            v4bf h;
            h.x = (__bf16)f.x; h.y = (__bf16)f.y; h.z = (__bf16)f.z; h.w = (__bf16)f.w;
            int lin = e * 256 + tid;
            *(v4bf*)&sA[(lin >> 3) * TK + a_kq[e] * 4] = h;
        }
        // ---- stage B tile transposed: gather a 4-deep column per chunk ----
        #pragma unroll
        for (int e = 0; e < 4; ++e) {
            int gcol = block_n + b_n[e];
            v4bf h;
            if (gcol < N) {
                const float* bp = &Bw[(long)(k0 + b_kq[e] * 4) * ldb + gcol];
                h.x = (__bf16)bp[0];
                h.y = (__bf16)bp[(long)ldb];
                h.z = (__bf16)bp[(long)ldb * 2];
                h.w = (__bf16)bp[(long)ldb * 3];
            } else {
                h.x = (__bf16)0.f; h.y = (__bf16)0.f; h.z = (__bf16)0.f; h.w = (__bf16)0.f;
            }
            *(v4bf*)&sB[b_n[e] * TK + b_kq[e] * 4] = h;
        }
        __syncthreads();

        // ---- fragment loads per ISA 16-bit A/B VGPR layout (ds_load_b128) ----
        ABfrag af[2], bf[4];
        #pragma unroll
        for (int im = 0; im < 2; ++im) {
            const __bf16* ap = &sA[(wm + im * 16 + mn) * TK];
            // lanes 0-15: K 0..7 then 16..23 ; lanes 16-31: K 8..15 then 24..31
            af[im].q[0] = *(const u32x4*)(ap + half * 8);
            af[im].q[1] = *(const u32x4*)(ap + 16 + half * 8);
        }
        #pragma unroll
        for (int in = 0; in < 4; ++in) {
            const __bf16* bp = &sB[(wn + in * 16 + mn) * TK];
            // lanes 0-15: K 0..7 then 16..23 ; lanes 16-31: K 8..15 then 24..31
            bf[in].q[0] = *(const u32x4*)(bp + half * 8);
            bf[in].q[1] = *(const u32x4*)(bp + 16 + half * 8);
        }
        #pragma unroll
        for (int im = 0; im < 2; ++im)
            #pragma unroll
            for (int in = 0; in < 4; ++in)
                acc[im][in] = __builtin_amdgcn_wmma_f32_16x16x32_bf16(
                    false, af[im].v, false, bf[in].v,
                    (short)0, acc[im][in], false, false);
        __syncthreads();
    }

    // ---- epilogue: C/D layout -> lanes 0-15 M=v, lanes 16-31 M=8+v ----
    #pragma unroll
    for (int im = 0; im < 2; ++im) {
        #pragma unroll
        for (int in = 0; in < 4; ++in) {
            int gcol = block_n + wn + in * 16 + mn;
            if (gcol >= N) continue;
            float bv = (mode >= 1) ? bias[gcol] : 0.0f;
            #pragma unroll
            for (int v = 0; v < 8; ++v) {
                int grow = block_m + wm + im * 16 + v + (half << 3);
                float val = acc[im][in][v] + bv;
                if (mode == 2) val = (val > 20.0f) ? val : __logf(1.0f + __expf(val));
                int orow = revC ? rev_row(grow) : grow;
                C[(long)orow * ldc + col_off + gcol] = val;
            }
        }
    }
}

// ---------------------------------------------------------------------------
// Depthwise causal conv (k=4) + bias + SiLU.  xi = xz[:, :DINNER].
// ---------------------------------------------------------------------------
__global__ __launch_bounds__(256)
void conv_silu_kernel(const float* __restrict__ xz, const float* __restrict__ cw,
                      const float* __restrict__ cb, float* __restrict__ xc,
                      int total, int ldxz)
{
    int idx = blockIdx.x * blockDim.x + threadIdx.x;
    if (idx >= total) return;
    int c = idx % DINNER;
    int l = (idx / DINNER) % SEQL;
    long row0 = (long)(idx / DINNER);
    float acc = cb[c];
    #pragma unroll
    for (int j = 0; j < DCONV; ++j) {
        int ls = l - (DCONV - 1) + j;
        if (ls >= 0) acc += xz[(row0 + (ls - l)) * ldxz + c] * cw[c * DCONV + j];
    }
    xc[idx] = silu_f(acc);
}

// ---------------------------------------------------------------------------
// Selective scan: one thread per (b, d); 16-wide state in registers.
// Fuses D-skip and SiLU(z) gating. B/C broadcast through LDS per timestep.
// ---------------------------------------------------------------------------
__global__ __launch_bounds__(256)
void scan_kernel(const float* __restrict__ dt, const float* __restrict__ dbc,
                 const float* __restrict__ xc, const float* __restrict__ xz,
                 const float* __restrict__ A_log, const float* __restrict__ Dsk,
                 float* __restrict__ y, int ldxz)
{
    __shared__ float sBC[2 * DSTATE];
    int gid = blockIdx.x * blockDim.x + threadIdx.x;
    int d = gid % DINNER;
    int b = gid / DINNER;          // whole block shares one b (2048 % 256 == 0)

    float Arow[DSTATE];
    #pragma unroll
    for (int n = 0; n < DSTATE; ++n) Arow[n] = -__expf(A_log[d * DSTATE + n]);
    const float Dd = Dsk[d];

    float h[DSTATE] = {};
    for (int l = 0; l < SEQL; ++l) {
        long row = (long)b * SEQL + l;
        __syncthreads();
        if (threadIdx.x < 2 * DSTATE)
            sBC[threadIdx.x] = dbc[row * 96 + DTRANK + threadIdx.x];
        __syncthreads();

        float dtv = dt[row * DINNER + d];
        float xv  = xc[row * DINNER + d];
        float zv  = xz[row * ldxz + DINNER + d];
        float dx  = dtv * xv;
        float ysum = 0.0f;
        #pragma unroll
        for (int n = 0; n < DSTATE; ++n) {
            float dA = __expf(dtv * Arow[n]);
            h[n] = h[n] * dA + dx * sBC[n];
            ysum += h[n] * sBC[DSTATE + n];
        }
        y[row * DINNER + d] = (ysum + xv * Dd) * silu_f(zv);
    }
}

// ---------------------------------------------------------------------------
static void run_gemm(const float* A, const float* Bw, float* C, const float* bias,
                     int M, int N, int K, int lda, int ldb, int ldc,
                     int col_off, int mode, int revA, int revC, hipStream_t s)
{
    dim3 grid((N + TN - 1) / TN, (M + TM - 1) / TM);
    gemm_bf16_wmma<<<grid, 256, 0, s>>>(A, Bw, C, bias, M, N, K, lda, ldb, ldc,
                                        col_off, mode, revA, revC);
}

extern "C" void kernel_launch(void* const* d_in, const int* in_sizes, int n_in,
                              void* d_out, int out_size, void* d_ws, size_t ws_size,
                              hipStream_t stream)
{
    const float* x      = (const float*)d_in[0];   // (B, L, 1024)
    const float* in_w   = (const float*)d_in[1];   // (2, 1024, 4096)
    const float* conv_w = (const float*)d_in[2];   // (2, 2048, 4)
    const float* conv_b = (const float*)d_in[3];   // (2, 2048)
    const float* xp_w   = (const float*)d_in[4];   // (2, 2048, 96)
    const float* dt_w   = (const float*)d_in[5];   // (2, 64, 2048)
    const float* dt_b   = (const float*)d_in[6];   // (2, 2048)
    const float* A_log  = (const float*)d_in[7];   // (2, 2048, 16)
    const float* Dsk    = (const float*)d_in[8];   // (2, 2048)
    const float* out_w  = (const float*)d_in[9];   // (2, 2048, 1024)
    const float* proj_w = (const float*)d_in[10];  // (2048, 1024)
    const float* proj_b = (const float*)d_in[11];  // (1024,)
    float* out = (float*)d_out;                    // (B, L, 1024)

    // Workspace layout (floats); ~101 MB total — fits in the 192MB global L2.
    float* xz  = (float*)d_ws;                          // (BL, 4096)
    float* xc  = xz  + (size_t)BL * 4096;               // (BL, 2048)
    float* dbc = xc  + (size_t)BL * DINNER;             // (BL, 96)
    float* dtb = dbc + (size_t)BL * 96;                 // (BL, 2048)
    float* yb  = dtb + (size_t)BL * DINNER;             // (BL, 2048)
    float* cat = yb  + (size_t)BL * DINNER;             // (BL, 2048)

    for (int dir = 0; dir < 2; ++dir) {
        int rev = (dir == 1);
        // 1) xz = x(dir-reversed) @ in_w[dir]           [2048 x 4096 x 1024]
        run_gemm(x, in_w + (size_t)dir * DMODEL * 2 * DINNER, xz, nullptr,
                 BL, 2 * DINNER, DMODEL, DMODEL, 2 * DINNER, 2 * DINNER,
                 0, 0, rev, 0, stream);
        // 2) xc = SiLU(depthwise_conv(xi) + b)
        {
            int total = BL * DINNER;
            conv_silu_kernel<<<(total + 255) / 256, 256, 0, stream>>>(
                xz, conv_w + (size_t)dir * DINNER * DCONV,
                conv_b + (size_t)dir * DINNER, xc, total, 2 * DINNER);
        }
        // 3) dbc = xc @ xp_w[dir]                       [2048 x 96 x 2048]
        run_gemm(xc, xp_w + (size_t)dir * DINNER * 96, dbc, nullptr,
                 BL, 96, DINNER, DINNER, 96, 96, 0, 0, 0, 0, stream);
        // 4) dt = softplus(dbc[:, :64] @ dt_w[dir] + dt_b[dir])
        run_gemm(dbc, dt_w + (size_t)dir * DTRANK * DINNER, dtb,
                 dt_b + (size_t)dir * DINNER,
                 BL, DINNER, DTRANK, 96, DINNER, DINNER, 0, 2, 0, 0, stream);
        // 5) selective scan (+ D skip + SiLU(z) gate)
        scan_kernel<<<(BSZ * DINNER) / 256, 256, 0, stream>>>(
            dtb, dbc, xc, xz,
            A_log + (size_t)dir * DINNER * DSTATE,
            Dsk + (size_t)dir * DINNER, yb, 2 * DINNER);
        // 6) cat[:, dir*1024 :] = y(@dir-reversed rows) @ out_w[dir]
        run_gemm(yb, out_w + (size_t)dir * DINNER * DMODEL, cat, nullptr,
                 BL, DMODEL, DINNER, DINNER, DMODEL, 2 * DMODEL,
                 dir * DMODEL, 0, 0, rev, stream);
    }
    // 7) out = cat @ proj_w + proj_b                    [2048 x 1024 x 2048]
    run_gemm(cat, proj_w, out, proj_b,
             BL, DMODEL, 2 * DMODEL, 2 * DMODEL, DMODEL, DMODEL,
             0, 1, 0, 0, stream);
}
